// FinePreprocess_996432413114
// MI455X (gfx1250) — compile-verified, matrix operand order checked by reading, not run on hardware
//
#include <hip/hip_runtime.h>
#include <hip/hip_bf16.h>

typedef __attribute__((ext_vector_type(16))) __bf16 v16bf;
typedef __attribute__((ext_vector_type(8)))  float  v8f;

#define DF   128
#define DC   256
#define HFD  208
#define WFD  208
#define WCC  104
#define LCC  (104 * 104)
#define NS   5000
#define WW   25     // 5x5 window

// workspace layout (bytes)
#define WS_CTX    0                       // 512 floats (f0 means | f1 means)
#define WS_WCF    2048                    // 12kt x 8nt fragments, bf16
#define WS_WMF    (WS_WCF + 12*8*32*16*2) // 8kt x 8nt
#define WS_WDP    (WS_WMF +  8*8*32*16*2) // 8kt x 8nt

// ---------------------------------------------------------------------------
// Kernel 1: spatial means of feat_f0 / feat_f1 -> ctx[2][2][128]
// ---------------------------------------------------------------------------
__global__ __launch_bounds__(256) void ctx_mean_kernel(const float* __restrict__ f0,
                                                       const float* __restrict__ f1,
                                                       float* __restrict__ ctx) {
    int bid = blockIdx.x;                 // 0..511 : [which][b*128+c]
    const float* src = (bid < 2 * DF) ? f0 : f1;
    int ch = bid & (2 * DF - 1);
    const float* p = src + (size_t)ch * (HFD * WFD);
    float s = 0.f;
    for (int i = threadIdx.x; i < HFD * WFD; i += 256) s += p[i];
    __shared__ float red[256];
    red[threadIdx.x] = s;
    __syncthreads();
    for (int off = 128; off > 0; off >>= 1) {
        if (threadIdx.x < off) red[threadIdx.x] += red[threadIdx.x + off];
        __syncthreads();
    }
    if (threadIdx.x == 0) ctx[bid] = red[0] * (1.0f / (float)(HFD * WFD));
}

// ---------------------------------------------------------------------------
// Kernel 2: pre-swizzle a row-major (ktiles*32 x 128) fp32 weight into bf16
// WMMA B-fragments: out[((kt*8+nt)*32 + lane)*16 + e], e per ISA K striping.
// ---------------------------------------------------------------------------
__global__ __launch_bounds__(256) void pack_b_kernel(const float* __restrict__ W,
                                                     int ktiles,
                                                     __bf16* __restrict__ out) {
    int t = blockIdx.x * 256 + threadIdx.x;      // one thread = one lane of one frag
    if (t >= ktiles * 8 * 32) return;
    int lane = t & 31;
    int frag = t >> 5;
    int kt = frag >> 3, nt = frag & 7;
    int n  = nt * 16 + (lane & 15);
    int hi = lane >> 4;
    __bf16* dst = out + (size_t)t * 16;
#pragma unroll
    for (int e = 0; e < 8; ++e) {
        dst[e]     = (__bf16)W[(size_t)(kt * 32 + hi * 8 + e) * DF + n];      // K lo run
        dst[e + 8] = (__bf16)W[(size_t)(kt * 32 + 16 + hi * 8 + e) * DF + n]; // K hi run
    }
}

// ---------------------------------------------------------------------------
// Fragment helpers (wave32 WMMA layouts per CDNA5 ISA 7.12.2)
// ---------------------------------------------------------------------------
__device__ __forceinline__ v16bf load_a_frag(const __bf16* __restrict__ base, int ld,
                                             int mtile, int ktile, int lane) {
    int m  = mtile * 16 + (lane & 15);
    int hi = lane >> 4;
    const __bf16* r = base + m * ld + ktile * 32 + hi * 8;
    v16bf a;
#pragma unroll
    for (int e = 0; e < 8; ++e) {
        a[e]     = r[e];        // K = hi*8 + e
        a[e + 8] = r[e + 16];   // K = 16 + hi*8 + e
    }
    return a;
}

__device__ __forceinline__ v16bf load_b_packed(const __bf16* __restrict__ packed,
                                               int kt, int nt, int lane) {
    return *(const v16bf*)(packed + ((size_t)((kt * 8 + nt) * 32 + lane)) * 16);
}

__device__ __forceinline__ v8f wmma_bf16(v16bf a, v16bf b, v8f c) {
    return __builtin_amdgcn_wmma_f32_16x16x32_bf16(false, a, false, b,
                                                   (short)0, c, false, false);
}

// ---------------------------------------------------------------------------
// Kernel 3: one block (8 waves) per sample. Produces out[n] and out[N+n].
// ---------------------------------------------------------------------------
__global__ __launch_bounds__(256) void fine_preprocess_kernel(
    const float* __restrict__ feat_f0, const float* __restrict__ feat_f1,
    const float* __restrict__ feat_c0, const float* __restrict__ feat_c1,
    const __bf16* __restrict__ packWcf, const float* __restrict__ bcf,
    const __bf16* __restrict__ packWdp, const float* __restrict__ bdp,
    const __bf16* __restrict__ packWmf, const float* __restrict__ bmf,
    const int* __restrict__ b_ids, const int* __restrict__ i_ids,
    const int* __restrict__ j_ids,
    const float* __restrict__ ctx,   // [0..255]=mean(f0), [256..511]=mean(f1)
    float* __restrict__ out)
{
    __shared__ __bf16 Acf[32][3 * DF];      // [c_patch | c_uav | c_map], padded M
    __shared__ __bf16 A2[2][32][2 * DF];    // [f | fc_rep] per pair (written late)
    // A_dp (16x256 down_proj input) aliases A2[0] rows 0..15: its lifetime ends
    // at the barrier before any A2 write.
    __bf16* A_dp = &A2[0][0][0];

    const int n   = blockIdx.x;
    const int tid = threadIdx.x;

    const int b  = b_ids[n];
    const int ii = i_ids[n];
    const int jj = j_ids[n];
    int x0 = (ii % WCC) * 2; x0 = x0 > WFD - 1 ? WFD - 1 : x0;
    int y0 = (ii / WCC) * 2; y0 = y0 > HFD - 1 ? HFD - 1 : y0;
    int x1 = (jj % WCC) * 2; x1 = x1 > WFD - 1 ? WFD - 1 : x1;
    int y1 = (jj / WCC) * 2; y1 = y1 > HFD - 1 ? HFD - 1 : y1;
    const float inv_s = 0.08838834764831845f;   // 1/sqrt(128)

    // ---- stage down_proj input tile: rows 0,1 = coarse rows, rows 2..15 = 0 ----
    {
        const float* r0 = feat_c0 + ((size_t)b * LCC + ii) * DC;
        const float* r1 = feat_c1 + ((size_t)b * LCC + jj) * DC;
        for (int idx = tid; idx < 16 * DC; idx += 256) {
            int r = idx >> 8;           // row 0..15
            int k = idx & (DC - 1);
            __bf16 v = (__bf16)0.0f;
            if (r == 0)      v = (__bf16)r0[k];
            else if (r == 1) v = (__bf16)r1[k];
            A_dp[idx] = v;
        }
    }

    // ---- build A = [c_patch | c_uav | c_map]  (rows 0..24; 25..31 zero) ----
    const int c = tid & (DF - 1);                 // constant per thread
    const float ctx0v = ctx[b * DF + c];          // mean(feat_f0)
    const float ctx1v = ctx[2 * DF + b * DF + c]; // mean(feat_f1)
    for (int idx = tid; idx < WW * DF; idx += 256) {
        int r  = idx >> 7;
        int dy = r / 5 - 2, dx = r % 5 - 2;
        int yy0 = y0 + dy, xx0 = x0 + dx;
        int yy1 = y1 + dy, xx1 = x1 + dx;
        float v0 = 0.f, v1 = 0.f;
        if (yy0 >= 0 && yy0 < HFD && xx0 >= 0 && xx0 < WFD)
            v0 = feat_f0[(((size_t)b * DF + c) * HFD + yy0) * WFD + xx0];
        if (yy1 >= 0 && yy1 < HFD && xx1 >= 0 && xx1 < WFD)
            v1 = feat_f1[(((size_t)b * DF + c) * HFD + yy1) * WFD + xx1];
        Acf[r][c]          = (__bf16)(v0 * v1 * inv_s);
        Acf[r][DF + c]     = (__bf16)(v0 * ctx1v * inv_s);
        Acf[r][2 * DF + c] = (__bf16)(v1 * ctx0v * inv_s);
    }
    for (int idx = tid; idx < 7 * 3 * DF; idx += 256)
        Acf[25 + idx / (3 * DF)][idx % (3 * DF)] = (__bf16)0.0f;
    __syncthreads();

    const int lane = tid & 31;
    const int nt   = tid >> 5;            // 0..7 : wave owns one N-tile
    const int ncol = nt * 16 + (lane & 15);
    const int hi   = lane >> 4;

    v8f zero = {};

    // ---- down_proj via WMMA: (16x256)@(256x128); valid rows 0,1 ----
    v8f accdp = zero;
    for (int kt = 0; kt < 8; ++kt) {
        v16bf bf = load_b_packed(packWdp, kt, nt, lane);
        v16bf a  = load_a_frag(A_dp, DC, 0, kt, lane);
        accdp = wmma_bf16(a, bf, accdp);
    }

    // ---- GEMM1: f{0,1} = A_perm @ Wcf ; B-fragment reused 4x ----
    v8f acc[2][2];
    acc[0][0] = zero; acc[0][1] = zero; acc[1][0] = zero; acc[1][1] = zero;
    for (int kt = 0; kt < 12; ++kt) {
        v16bf bf = load_b_packed(packWcf, kt, nt, lane);
        // pair 0: natural K order; pair 1 swaps c_uav <-> c_map blocks
        int akt1 = kt < 4 ? kt : (kt < 8 ? kt + 4 : kt - 4);
#pragma unroll
        for (int mt = 0; mt < 2; ++mt) {
            v16bf a0 = load_a_frag(&Acf[0][0], 3 * DF, mt, kt, lane);
            acc[0][mt] = wmma_bf16(a0, bf, acc[0][mt]);
            v16bf a1 = load_a_frag(&Acf[0][0], 3 * DF, mt, akt1, lane);
            acc[1][mt] = wmma_bf16(a1, bf, acc[1][mt]);
        }
    }
    __syncthreads();   // all A_dp/Acf reads done; A2 may now be overwritten

    // fc columns: rows 0 (fc0) and 1 (fc1) of the dp tile live in hi==0 lanes
    if (hi == 0) {
        float dpb = bdp[ncol];
        __bf16 h0 = (__bf16)(accdp[0] + dpb);
        __bf16 h1 = (__bf16)(accdp[1] + dpb);
        for (int r = 0; r < 32; ++r) {
            A2[0][r][DF + ncol] = h0;
            A2[1][r][DF + ncol] = h1;
        }
    }
    // f columns + bias per ISA C/D layout
    const float cb = bcf[ncol];
#pragma unroll
    for (int pair = 0; pair < 2; ++pair)
#pragma unroll
        for (int mt = 0; mt < 2; ++mt)
#pragma unroll
            for (int v = 0; v < 8; ++v) {
                int row = mt * 16 + v + 8 * hi;
                A2[pair][row][ncol] = (__bf16)(acc[pair][mt][v] + cb);
            }
    __syncthreads();

    // ---- GEMM2: merged = A2 @ Wmf + bmf ----
    const float mb = bmf[ncol];
    v8f acc2[2][2];
    acc2[0][0] = zero; acc2[0][1] = zero; acc2[1][0] = zero; acc2[1][1] = zero;
    for (int kt = 0; kt < 8; ++kt) {
        v16bf bf = load_b_packed(packWmf, kt, nt, lane);
#pragma unroll
        for (int pair = 0; pair < 2; ++pair)
#pragma unroll
            for (int mt = 0; mt < 2; ++mt) {
                v16bf a = load_a_frag(&A2[pair][0][0], 2 * DF, mt, kt, lane);
                acc2[pair][mt] = wmma_bf16(a, bf, acc2[pair][mt]);
            }
    }

#pragma unroll
    for (int pair = 0; pair < 2; ++pair) {
        size_t base = (size_t)(pair ? NS + n : n) * WW * DF;
#pragma unroll
        for (int mt = 0; mt < 2; ++mt)
#pragma unroll
            for (int v = 0; v < 8; ++v) {
                int row = mt * 16 + v + 8 * hi;
                if (row < WW)
                    out[base + (size_t)row * DF + ncol] = acc2[pair][mt][v] + mb;
            }
    }
}

extern "C" void kernel_launch(void* const* d_in, const int* in_sizes, int n_in,
                              void* d_out, int out_size, void* d_ws, size_t ws_size,
                              hipStream_t stream) {
    const float* feat_f0 = (const float*)d_in[0];
    const float* feat_f1 = (const float*)d_in[1];
    const float* feat_c0 = (const float*)d_in[2];
    const float* feat_c1 = (const float*)d_in[3];
    const float* Wcf     = (const float*)d_in[4];
    const float* bcf     = (const float*)d_in[5];
    const float* Wdp     = (const float*)d_in[6];
    const float* bdp     = (const float*)d_in[7];
    const float* Wmf     = (const float*)d_in[8];
    const float* bmf     = (const float*)d_in[9];
    const int*   b_ids   = (const int*)d_in[10];
    const int*   i_ids   = (const int*)d_in[11];
    const int*   j_ids   = (const int*)d_in[12];
    // d_in[13], d_in[14] = w0c, w1c (compile-time constants = 104)

    char* ws = (char*)d_ws;
    float*  ctx     = (float*)(ws + WS_CTX);
    __bf16* packWcf = (__bf16*)(ws + WS_WCF);
    __bf16* packWmf = (__bf16*)(ws + WS_WMF);
    __bf16* packWdp = (__bf16*)(ws + WS_WDP);

    ctx_mean_kernel<<<512, 256, 0, stream>>>(feat_f0, feat_f1, ctx);
    pack_b_kernel<<<(12 * 8 * 32 + 255) / 256, 256, 0, stream>>>(Wcf, 12, packWcf);
    pack_b_kernel<<<( 8 * 8 * 32 + 255) / 256, 256, 0, stream>>>(Wmf,  8, packWmf);
    pack_b_kernel<<<( 8 * 8 * 32 + 255) / 256, 256, 0, stream>>>(Wdp,  8, packWdp);

    fine_preprocess_kernel<<<NS, 256, 0, stream>>>(
        feat_f0, feat_f1, feat_c0, feat_c1,
        packWcf, bcf, packWdp, bdp, packWmf, bmf,
        b_ids, i_ids, j_ids, ctx, (float*)d_out);
}
